// GAT_custom_hidden_47562467836510
// MI455X (gfx1250) — matrix operand order repaired and checked by reading, not compiled, not run on hardware
//
#include <hip/hip_runtime.h>
#include <hip/hip_bf16.h>

typedef __attribute__((ext_vector_type(16))) _Float16 v16h;
typedef __attribute__((ext_vector_type(8)))  _Float16 v8h;
typedef __attribute__((ext_vector_type(8)))  float    v8f;

#define NEG_SLOPE 0.2f
#define BN_EPS    1e-5f

// ---------------- ordered-float encoding for atomic max on f32 (handles negatives) ---------
__device__ __forceinline__ unsigned encf(float f) {
    unsigned u = __float_as_uint(f);
    return (u & 0x80000000u) ? ~u : (u | 0x80000000u);
}
__device__ __forceinline__ float decf(unsigned u) {
    return (u & 0x80000000u) ? __uint_as_float(u & 0x7fffffffu) : __uint_as_float(~u);
}
#define ENC_NEG_INF 0x007FFFFFu   // encf(-inf)

__device__ __forceinline__ float lrelu(float v) { return v > 0.f ? v : NEG_SLOPE * v; }

__device__ __forceinline__ v16h cat16(v8h lo, v8h hi) {
    v16h r;
#pragma unroll
    for (int i = 0; i < 8; ++i) { r[i] = lo[i]; r[8 + i] = hi[i]; }
    return r;
}

// ---------------- operand pre-conversion -------------------------------------------------
// Wt[m][k] = (f16) W[k][m]  (transpose so B fragments are contiguous along K)
__global__ void w_cvt_t(const float* __restrict__ W, _Float16* __restrict__ Wt, int K, int M) {
    int i = blockIdx.x * blockDim.x + threadIdx.x;
    if (i >= K * M) return;
    int k = i / M, m = i % M;
    Wt[(size_t)m * K + k] = (_Float16)W[i];
}

__global__ void x_cvt(const float* __restrict__ X, _Float16* __restrict__ Xh, long long total) {
    long long i = (long long)blockIdx.x * blockDim.x + threadIdx.x;
    if (i < total) Xh[i] = (_Float16)X[i];
}

// ---------------- WMMA GEMM: Out[N,M] = Xh[N,K] * Wt[M,K]^T, f16 in, f32 accum ------------
// One wave computes a 16 x (16*MT) slab: A fragment reused across MT B fragments.
// All operand loads are contiguous 16-byte global_load_b128.
template <int K, int M, int MT>
__global__ __launch_bounds__(32) void gemm_wmma_f16(const _Float16* __restrict__ Xh,
                                                    const _Float16* __restrict__ Wt,
                                                    float* __restrict__ Out) {
    const int groupsM = M / (16 * MT);
    const int tmg = (blockIdx.x % groupsM) * 16 * MT;  // first output-feature column of slab
    const int tn  = (blockIdx.x / groupsM) * 16;       // node-row tile
    const int lane = threadIdx.x;                      // 0..31
    const int half = lane >> 4;                        // 0 | 1
    const int l    = lane & 15;

    v8f acc[MT];
#pragma unroll
    for (int j = 0; j < MT; ++j) acc[j] = (v8f){};

#pragma unroll
    for (int kb = 0; kb < K; kb += 32) {
        // A fragment (16x32 f16): row = tn+l.
        // lane<16 : halves 0..7 -> K=kb+0..7,  halves 8..15 -> K=kb+16..23
        // lane>=16: halves 0..7 -> K=kb+8..15, halves 8..15 -> K=kb+24..31
        const _Float16* xr = Xh + (size_t)(tn + l) * K + kb + (half ? 8 : 0);
        v16h a = cat16(*(const v8h*)xr, *(const v8h*)(xr + 16));

#pragma unroll
        for (int j = 0; j < MT; ++j) {
            // B fragment (32x16 f16): col = tmg + 16j + l,
            // lanes<16 K=kb+0..7,kb+16..23, lanes>=16 K=kb+8..15,kb+24..31 (same order as A).
            const _Float16* wr = Wt + (size_t)(tmg + j * 16 + l) * K + kb + (half ? 8 : 0);
            v16h b = cat16(*(const v8h*)wr, *(const v8h*)(wr + 16));
            acc[j] = __builtin_amdgcn_wmma_f32_16x16x32_f16(false, a, false, b,
                                                            (short)0, acc[j], false, false);
        }
    }
    // D store: VGPR r -> row (tn + r + 8*half), col (tmg + 16j + l)
#pragma unroll
    for (int j = 0; j < MT; ++j) {
        float* orow = Out + (size_t)(tn + half * 8) * M + tmg + j * 16 + l;
#pragma unroll
        for (int r = 0; r < 8; ++r) orow[(size_t)r * M] = acc[j][r];
    }
}

// ---------------- per-(node,head) attention scores: s = <h, a_src>, d = <h, a_dst> ---------
__global__ void attn_scores(const float* __restrict__ H, const float* __restrict__ a_src,
                            const float* __restrict__ a_dst, float* __restrict__ s_nd,
                            float* __restrict__ d_nd, int N, int Hh, int C) {
    int idx = blockIdx.x * blockDim.x + threadIdx.x;
    if (idx >= N * Hh) return;
    int n = idx / Hh, h = idx % Hh;
    const float* hr = H + (size_t)n * Hh * C + (size_t)h * C;
    const float* as = a_src + (size_t)h * C;
    const float* ad = a_dst + (size_t)h * C;
    float ss = 0.f, dd = 0.f;
    for (int c = 0; c < C; ++c) { float v = hr[c]; ss += v * as[c]; dd += v * ad[c]; }
    s_nd[idx] = ss; d_nd[idx] = dd;
}

__global__ void init_attn(unsigned* __restrict__ emax, float* __restrict__ esum, int n) {
    int i = blockIdx.x * blockDim.x + threadIdx.x;
    if (i < n) { emax[i] = ENC_NEG_INF; esum[i] = 0.f; }
}

__global__ void init_out(float* __restrict__ out, const float* __restrict__ b, long long total, int M) {
    long long i = (long long)blockIdx.x * blockDim.x + threadIdx.x;
    if (i < total) out[i] = b[i % M];
}

__device__ __forceinline__ void edge_ends(const int* ei, int e, int NE, int& src, int& dst) {
    if (e < NE) { src = ei[e]; dst = ei[NE + e]; }
    else        { src = e - NE; dst = e - NE; }        // self loops appended
}

// ---------------- segment max over incoming edges (ordered-uint atomicMax) -----------------
__global__ void edge_max_k(const int* __restrict__ ei, const float* __restrict__ s_nd,
                           const float* __restrict__ d_nd, unsigned* __restrict__ emax,
                           int NE, int N, int Hh) {
    long long idx = (long long)blockIdx.x * blockDim.x + threadIdx.x;
    if (idx >= (long long)(NE + N) * Hh) return;
    int e = (int)(idx / Hh), h = (int)(idx % Hh);
    int src, dst; edge_ends(ei, e, NE, src, dst);
    float v = lrelu(s_nd[(size_t)src * Hh + h] + d_nd[(size_t)dst * Hh + h]);
    atomicMax(&emax[(size_t)dst * Hh + h], encf(v));
}

// ---------------- exp(e - max) and segment-sum denominator ---------------------------------
__global__ void edge_exp_k(const int* __restrict__ ei, const float* __restrict__ s_nd,
                           const float* __restrict__ d_nd, const unsigned* __restrict__ emax,
                           float* __restrict__ esum, float* __restrict__ ealpha,
                           int NE, int N, int Hh) {
    long long idx = (long long)blockIdx.x * blockDim.x + threadIdx.x;
    if (idx >= (long long)(NE + N) * Hh) return;
    int e = (int)(idx / Hh), h = (int)(idx % Hh);
    int src, dst; edge_ends(ei, e, NE, src, dst);
    float v = lrelu(s_nd[(size_t)src * Hh + h] + d_nd[(size_t)dst * Hh + h]);
    float w = __expf(v - decf(emax[(size_t)dst * Hh + h]));
    ealpha[idx] = w;
    atomicAdd(&esum[(size_t)dst * Hh + h], w);
}

// ---------------- weighted gather-scatter aggregation: out[dst] += alpha * h[src] ----------
// 4 channels per thread (float4 gather), M/4 threads per edge.
__global__ void edge_agg_k(const int* __restrict__ ei, const float* __restrict__ Hfeat,
                           const float* __restrict__ ealpha, const float* __restrict__ esum,
                           float* __restrict__ out, int NE, int N, int Hh, int C) {
    const int M = Hh * C;
    const int tpe = M / 4;
    long long gid = (long long)blockIdx.x * blockDim.x + threadIdx.x;
    if (gid >= (long long)(NE + N) * tpe) return;
    int e  = (int)(gid / tpe);
    int c0 = (int)(gid % tpe) * 4;
    int h  = c0 / C;
    int src, dst; edge_ends(ei, e, NE, src, dst);
    float alpha = ealpha[(size_t)e * Hh + h] / esum[(size_t)dst * Hh + h];
    float4 hv = *(const float4*)(Hfeat + (size_t)src * M + c0);
    float* o = out + (size_t)dst * M + c0;
    atomicAdd(o + 0, hv.x * alpha);
    atomicAdd(o + 1, hv.y * alpha);
    atomicAdd(o + 2, hv.z * alpha);
    atomicAdd(o + 3, hv.w * alpha);
}

// ---------------- fused batchnorm (inference) + relu; writes f32 and f16 copies ------------
__global__ void bn_relu_k(float* __restrict__ x, _Float16* __restrict__ xh,
                          const float* __restrict__ g, const float* __restrict__ be,
                          const float* __restrict__ m, const float* __restrict__ v,
                          long long total, int M) {
    long long i = (long long)blockIdx.x * blockDim.x + threadIdx.x;
    if (i >= total) return;
    int c = (int)(i % M);
    float y = (x[i] - m[c]) * rsqrtf(v[c] + BN_EPS) * g[c] + be[c];
    y = y > 0.f ? y : 0.f;
    x[i] = y;
    xh[i] = (_Float16)y;
}

static inline dim3 g1(long long t) { return dim3((unsigned)((t + 255) / 256)); }

extern "C" void kernel_launch(void* const* d_in, const int* in_sizes, int n_in,
                              void* d_out, int out_size, void* d_ws, size_t ws_size,
                              hipStream_t stream) {
    const int N = 50000, E = 800000, ET = E + N;
    const float* x   = (const float*)d_in[0];
    const int*   ei  = (const int*)d_in[1];
    const float* W1  = (const float*)d_in[2];
    const float* as1 = (const float*)d_in[3];
    const float* ad1 = (const float*)d_in[4];
    const float* b1  = (const float*)d_in[5];
    const float* gm1 = (const float*)d_in[6];
    const float* be1 = (const float*)d_in[7];
    const float* mu1 = (const float*)d_in[8];
    const float* va1 = (const float*)d_in[9];
    const float* W2  = (const float*)d_in[10];
    const float* as2 = (const float*)d_in[11];
    const float* ad2 = (const float*)d_in[12];
    const float* b2  = (const float*)d_in[13];
    const float* gm2 = (const float*)d_in[14];
    const float* be2 = (const float*)d_in[15];
    const float* mu2 = (const float*)d_in[16];
    const float* va2 = (const float*)d_in[17];
    const float* W3  = (const float*)d_in[18];
    const float* as3 = (const float*)d_in[19];
    const float* ad3 = (const float*)d_in[20];
    const float* b3  = (const float*)d_in[21];
    float* out = (float*)d_out;

    // workspace carve-up (~145 MB)
    float*    hbuf   = (float*)d_ws;                        // N*256 f32
    float*    obuf   = hbuf + (size_t)N * 256;              // N*256 f32
    float*    s_nd   = obuf + (size_t)N * 256;              // N*4
    float*    d_nd   = s_nd + (size_t)N * 4;                // N*4
    unsigned* emax   = (unsigned*)(d_nd + (size_t)N * 4);   // N*4
    float*    esum   = (float*)(emax + (size_t)N * 4);      // N*4
    float*    ealpha = esum + (size_t)N * 4;                // ET*4
    _Float16* xh     = (_Float16*)(ealpha + (size_t)ET * 4);// N*256 f16
    _Float16* wt     = xh + (size_t)N * 256;                // 256*256 f16

    const dim3 blk(256);
    const int tilesN = N / 16;   // 3125

    // =================== Layer 1: GATConv(128 -> 64, heads=4, concat) ===================
    w_cvt_t<<<g1(128 * 256), blk, 0, stream>>>(W1, wt, 128, 256);
    x_cvt<<<g1((long long)N * 128), blk, 0, stream>>>(x, xh, (long long)N * 128);
    gemm_wmma_f16<128, 256, 4><<<dim3(tilesN * 4), dim3(32), 0, stream>>>(xh, wt, hbuf);
    attn_scores<<<g1((long long)N * 4), blk, 0, stream>>>(hbuf, as1, ad1, s_nd, d_nd, N, 4, 64);
    init_attn<<<g1((long long)N * 4), blk, 0, stream>>>(emax, esum, N * 4);
    init_out<<<g1((long long)N * 256), blk, 0, stream>>>(obuf, b1, (long long)N * 256, 256);
    edge_max_k<<<g1((long long)ET * 4), blk, 0, stream>>>(ei, s_nd, d_nd, emax, E, N, 4);
    edge_exp_k<<<g1((long long)ET * 4), blk, 0, stream>>>(ei, s_nd, d_nd, emax, esum, ealpha, E, N, 4);
    edge_agg_k<<<g1((long long)ET * 64), blk, 0, stream>>>(ei, hbuf, ealpha, esum, obuf, E, N, 4, 64);
    bn_relu_k<<<g1((long long)N * 256), blk, 0, stream>>>(obuf, xh, gm1, be1, mu1, va1, (long long)N * 256, 256);

    // =================== Layer 2: GATConv(256 -> 64, heads=4, concat) ===================
    w_cvt_t<<<g1(256 * 256), blk, 0, stream>>>(W2, wt, 256, 256);
    gemm_wmma_f16<256, 256, 4><<<dim3(tilesN * 4), dim3(32), 0, stream>>>(xh, wt, hbuf);
    attn_scores<<<g1((long long)N * 4), blk, 0, stream>>>(hbuf, as2, ad2, s_nd, d_nd, N, 4, 64);
    init_attn<<<g1((long long)N * 4), blk, 0, stream>>>(emax, esum, N * 4);
    init_out<<<g1((long long)N * 256), blk, 0, stream>>>(obuf, b2, (long long)N * 256, 256);
    edge_max_k<<<g1((long long)ET * 4), blk, 0, stream>>>(ei, s_nd, d_nd, emax, E, N, 4);
    edge_exp_k<<<g1((long long)ET * 4), blk, 0, stream>>>(ei, s_nd, d_nd, emax, esum, ealpha, E, N, 4);
    edge_agg_k<<<g1((long long)ET * 64), blk, 0, stream>>>(ei, hbuf, ealpha, esum, obuf, E, N, 4, 64);
    bn_relu_k<<<g1((long long)N * 256), blk, 0, stream>>>(obuf, xh, gm2, be2, mu2, va2, (long long)N * 256, 256);

    // =================== Layer 3: GATConv(256 -> 16, heads=1, no concat) ================
    w_cvt_t<<<g1(256 * 16), blk, 0, stream>>>(W3, wt, 256, 16);
    gemm_wmma_f16<256, 16, 1><<<dim3(tilesN), dim3(32), 0, stream>>>(xh, wt, hbuf);
    attn_scores<<<g1((long long)N), blk, 0, stream>>>(hbuf, as3, ad3, s_nd, d_nd, N, 1, 16);
    init_attn<<<g1((long long)N), blk, 0, stream>>>(emax, esum, N);
    init_out<<<g1((long long)N * 16), blk, 0, stream>>>(out, b3, (long long)N * 16, 16);
    edge_max_k<<<g1((long long)ET), blk, 0, stream>>>(ei, s_nd, d_nd, emax, E, N, 1);
    edge_exp_k<<<g1((long long)ET), blk, 0, stream>>>(ei, s_nd, d_nd, emax, esum, ealpha, E, N, 1);
    edge_agg_k<<<g1((long long)ET * 4), blk, 0, stream>>>(ei, hbuf, ealpha, esum, out, E, N, 1, 16);
}
